// _MambaBlock_72825465471238
// MI455X (gfx1250) — matrix operand order repaired and checked by reading, not compiled, not run on hardware
//
#include <hip/hip_runtime.h>
#include <hip/hip_bf16.h>

// ---------------------------------------------------------------------------
// MambaBlock on MI455X (gfx1250): bf16 WMMA GEMMs + O(L) recurrence scan.
//   B=8, L=512, DIM=256, INNER=512, M = B*L = 4096
// ---------------------------------------------------------------------------

#define BB     8
#define LL     512
#define DIMX   256
#define INNERX 512
#define MROWS  (BB * LL)          // 4096
#define NP     (2 * INNERX)      // 1024 (GEMM1 output cols)

typedef __attribute__((ext_vector_type(16))) __bf16        v16bf;
typedef __attribute__((ext_vector_type(8)))  float         v8f;
typedef __attribute__((ext_vector_type(4)))  unsigned int  u32x4;
typedef __attribute__((ext_vector_type(8)))  unsigned int  u32x8;

static __device__ __forceinline__ unsigned short f32_to_bf16(float f) {
    unsigned int u = __float_as_uint(f);
    unsigned int r = u + 0x7FFFu + ((u >> 16) & 1u);   // round-to-nearest-even
    return (unsigned short)(r >> 16);
}

static __device__ __forceinline__ float sigmoidf_fast(float v) {
    return 1.0f / (1.0f + __expf(-v));
}

// A fragment (16x32 bf16, M x K), row-major source [*, row_stride] (bf16):
//   lane l -> row (row0 + (l&15)); dwords0-3: K = k0 + (l>>4)*8 + {0..7}
//                                  dwords4-7: K = that + 16
static __device__ __forceinline__ v16bf
load_a_frag(const unsigned short* __restrict__ base, int row_stride,
            int lane, int row0, int k0) {
    const int r  = row0 + (lane & 15);
    const int kb = k0 + ((lane >> 4) << 3);
    const unsigned short* p = base + (size_t)r * row_stride + kb;
    u32x4 lo = *(const u32x4*)(p);
    u32x4 hi = *(const u32x4*)(p + 16);
    u32x8 t  = __builtin_shufflevector(lo, hi, 0, 1, 2, 3, 4, 5, 6, 7);
    return __builtin_bit_cast(v16bf, t);
}

// B fragment (32x16 bf16, K x N), sourced from B^T stored row-major [N][K]:
//   lane l -> col (n0 + (l&15)); dwords0-3: K = k0 + (l>>4)*8 + {0..7}, dwords4-7: K = that + 16
static __device__ __forceinline__ v16bf
load_b_frag(const unsigned short* __restrict__ baseT, int row_stride,
            int lane, int n0, int k0) {
    const int n  = n0 + (lane & 15);
    const int kb = k0 + ((lane >> 4) << 3);
    const unsigned short* p = baseT + (size_t)n * row_stride + kb;
    u32x4 lo = *(const u32x4*)(p);
    u32x4 hi = *(const u32x4*)(p + 16);
    u32x8 t  = __builtin_shufflevector(lo, hi, 0, 1, 2, 3, 4, 5, 6, 7);
    return __builtin_bit_cast(v16bf, t);
}

static __device__ __forceinline__ v8f
wmma_bf16(v16bf a, v16bf b, v8f c) {
    return __builtin_amdgcn_wmma_f32_16x16x32_bf16(
        /*neg_a=*/false, a, /*neg_b=*/false, b,
        /*c_mod=*/(short)0, c, /*reuse_a=*/false, /*reuse_b=*/false);
}

// ---------------------------------------------------------------------------
// Kernel 0: f32 -> bf16 conversion (grid-stride)
// ---------------------------------------------------------------------------
__global__ void cvt_bf16_kernel(const float* __restrict__ in,
                                unsigned short* __restrict__ out, int n) {
    for (int i = blockIdx.x * blockDim.x + threadIdx.x; i < n;
         i += gridDim.x * blockDim.x)
        out[i] = f32_to_bf16(in[i]);
}

// ---------------------------------------------------------------------------
// Kernel 1: GEMM1  xp[4096,1024] = x_bf16[4096,256] . Win_bf16[1024,256]^T
//   cols [0,512)  -> u_buf (f32)
//   cols [512,1024) -> g_buf = sigmoid(xp) (f32)
// Block: 256 threads = 8 waves arranged 4(M) x 2(N); wave tile 16x64.
// Grid: (4096/64, 1024/128) = (64, 8)
// ---------------------------------------------------------------------------
__global__ void __launch_bounds__(256)
gemm1_kernel(const unsigned short* __restrict__ xb,     // [4096][256]
             const unsigned short* __restrict__ winb,   // [1024][256]
             float* __restrict__ u_buf,                 // [4096][512]
             float* __restrict__ g_buf) {               // [4096][512]
    const int lane = threadIdx.x & 31;
    const int wave = threadIdx.x >> 5;
    const int wm   = wave & 3;
    const int wn   = wave >> 2;
    const int m0   = blockIdx.x * 64 + wm * 16;
    const int n0   = blockIdx.y * 128 + wn * 64;

    v8f acc[4] = {};
    #pragma unroll
    for (int k = 0; k < DIMX; k += 32) {
        if (k + 32 < DIMX)
            __builtin_prefetch(xb + (size_t)(m0 + (lane & 15)) * DIMX + k + 32, 0, 1);
        v16bf a = load_a_frag(xb, DIMX, lane, m0, k);
        #pragma unroll
        for (int t = 0; t < 4; ++t) {
            v16bf b = load_b_frag(winb, DIMX, lane, n0 + t * 16, k);
            acc[t] = wmma_bf16(a, b, acc[t]);
        }
    }

    const int mb = m0 + ((lane >> 4) << 3);
    #pragma unroll
    for (int t = 0; t < 4; ++t) {
        const int n = n0 + t * 16 + (lane & 15);
        if (n < INNERX) {
            #pragma unroll
            for (int i = 0; i < 8; ++i)
                u_buf[(size_t)(mb + i) * INNERX + n] = acc[t][i];
        } else {
            const int ng = n - INNERX;
            #pragma unroll
            for (int i = 0; i < 8; ++i)
                g_buf[(size_t)(mb + i) * INNERX + ng] = sigmoidf_fast(acc[t][i]);
        }
    }
}

// ---------------------------------------------------------------------------
// Kernel 2: depthwise causal conv (k=3) + first-order scan + gate, per
// (batch, channel). The reference's O(L^2) "M" einsum == y[t] = d*y[t-1]+inp[t].
// 4096 threads, coalesced over channel dim. Emits bf16 y for GEMM2.
// ---------------------------------------------------------------------------
__global__ void __launch_bounds__(256)
conv_scan_gate_kernel(const float* __restrict__ u_buf,        // [B][L][INNER]
                      const float* __restrict__ g_buf,        // sigmoid(g)
                      const float* __restrict__ conv_w,       // [INNER][1][3]
                      const float* __restrict__ state_decay,  // [INNER]
                      const float* __restrict__ state_scale,  // [INNER]
                      unsigned short* __restrict__ y_bf) {    // [B][L][INNER]
    const int tid = blockIdx.x * blockDim.x + threadIdx.x;   // 0..4095
    const int dch = tid & (INNERX - 1);
    const int b   = tid >> 9;

    const float w0 = conv_w[dch * 3 + 0];
    const float w1 = conv_w[dch * 3 + 1];
    const float w2 = conv_w[dch * 3 + 2];
    const float dd   = sigmoidf_fast(state_decay[dch]);   // log_d path: d^(t-k)
    const float coef = (1.0f - dd) * state_scale[dch];

    const float* up = u_buf + (size_t)b * LL * INNERX + dch;
    const float* gp = g_buf + (size_t)b * LL * INNERX + dch;
    unsigned short* yp = y_bf + (size_t)b * LL * INNERX + dch;

    float um2 = 0.0f, um1 = 0.0f, y = 0.0f;
    for (int t = 0; t < LL; ++t) {
        const float uc = up[(size_t)t * INNERX];
        const float c  = w0 * um2 + w1 * um1 + w2 * uc;   // left-pad-2 causal
        um2 = um1; um1 = uc;
        y = dd * y + coef * c;                            // linear recurrence
        yp[(size_t)t * INNERX] = f32_to_bf16(y * gp[(size_t)t * INNERX]);
    }
}

// ---------------------------------------------------------------------------
// Kernel 3: GEMM2 out[4096,256] = y_bf16[4096,512] . Wout_bf16[256,512]^T
// fused with RMSNorm over the 256-wide output row.
// Block: 256 threads = 8 waves; block owns a full 16x256 output slab
// (wave w -> N columns [w*32, w*32+32)). Grid: 4096/16 = 256 blocks.
// ---------------------------------------------------------------------------
__global__ void __launch_bounds__(256)
gemm2_norm_kernel(const unsigned short* __restrict__ yb,      // [4096][512]
                  const unsigned short* __restrict__ woutb,   // [256][512]
                  const float* __restrict__ norm_w,           // [256]
                  float* __restrict__ out) {                  // [4096][256]
    __shared__ float tileY[16][DIMX + 16];
    __shared__ float psum[16][16];
    __shared__ float rstd[16];

    const int lane = threadIdx.x & 31;
    const int wave = threadIdx.x >> 5;
    const int m0   = blockIdx.x * 16;
    const int nw   = wave * 32;

    v8f acc[2] = {};
    #pragma unroll
    for (int k = 0; k < INNERX; k += 32) {
        if (k + 32 < INNERX)
            __builtin_prefetch(yb + (size_t)(m0 + (lane & 15)) * INNERX + k + 32, 0, 1);
        v16bf a = load_a_frag(yb, INNERX, lane, m0, k);
        #pragma unroll
        for (int t = 0; t < 2; ++t) {
            v16bf b = load_b_frag(woutb, INNERX, lane, nw + t * 16, k);
            acc[t] = wmma_bf16(a, b, acc[t]);
        }
    }

    const int rb = (lane >> 4) << 3;          // local row base (0 or 8)
    #pragma unroll
    for (int t = 0; t < 2; ++t) {
        const int n = nw + t * 16 + (lane & 15);
        #pragma unroll
        for (int i = 0; i < 8; ++i)
            tileY[rb + i][n] = acc[t][i];
    }
    __syncthreads();

    // RMSNorm: 16 threads per row, 16 cols per thread.
    const int r   = threadIdx.x >> 4;
    const int seg = threadIdx.x & 15;
    float s = 0.0f;
    #pragma unroll
    for (int j = 0; j < 16; ++j) {
        const float v = tileY[r][seg * 16 + j];
        s += v * v;
    }
    psum[r][seg] = s;
    __syncthreads();
    if (threadIdx.x < 16) {
        float ms = 0.0f;
        #pragma unroll
        for (int j = 0; j < 16; ++j) ms += psum[threadIdx.x][j];
        rstd[threadIdx.x] = rsqrtf(ms * (1.0f / (float)DIMX) + 1e-6f);
    }
    __syncthreads();

    const float rs = rstd[r];
    #pragma unroll
    for (int j = 0; j < 16; ++j) {
        const int c = seg * 16 + j;
        out[(size_t)(m0 + r) * DIMX + c] = tileY[r][c] * rs * norm_w[c];
    }
}

// ---------------------------------------------------------------------------
// Workspace layout (bytes, 256-aligned)
// ---------------------------------------------------------------------------
#define OFF_XBF    0u                                  // 4096*256*2  = 2 MiB
#define OFF_WINBF  (OFF_XBF   + MROWS * DIMX * 2u)     // 1024*256*2  = 512 KiB
#define OFF_WOUTBF (OFF_WINBF + NP * DIMX * 2u)        // 256*512*2   = 256 KiB
#define OFF_UBUF   (OFF_WOUTBF + DIMX * INNERX * 2u)   // 4096*512*4  = 8 MiB
#define OFF_GBUF   (OFF_UBUF  + MROWS * INNERX * 4u)   // 4096*512*4  = 8 MiB
#define OFF_YBF    (OFF_GBUF  + MROWS * INNERX * 4u)   // 4096*512*2  = 4 MiB

extern "C" void kernel_launch(void* const* d_in, const int* in_sizes, int n_in,
                              void* d_out, int out_size, void* d_ws, size_t ws_size,
                              hipStream_t stream) {
    const float* x           = (const float*)d_in[0];  // [8,512,256]
    const float* Win         = (const float*)d_in[1];  // [1024,256]
    const float* conv_w      = (const float*)d_in[2];  // [512,1,3]
    const float* state_decay = (const float*)d_in[3];  // [512]
    const float* state_scale = (const float*)d_in[4];  // [512]
    const float* Wout        = (const float*)d_in[5];  // [256,512]
    const float* norm_w      = (const float*)d_in[6];  // [256]
    float* out = (float*)d_out;                        // [8,512,256]

    char* ws = (char*)d_ws;
    unsigned short* x_bf    = (unsigned short*)(ws + OFF_XBF);
    unsigned short* win_bf  = (unsigned short*)(ws + OFF_WINBF);
    unsigned short* wout_bf = (unsigned short*)(ws + OFF_WOUTBF);
    float*          u_buf   = (float*)(ws + OFF_UBUF);
    float*          g_buf   = (float*)(ws + OFF_GBUF);
    unsigned short* y_bf    = (unsigned short*)(ws + OFF_YBF);

    // 0) f32 -> bf16 conversions
    {
        int n0 = MROWS * DIMX;   // x
        int n1 = NP * DIMX;      // Win
        int n2 = DIMX * INNERX;  // Wout
        cvt_bf16_kernel<<<(n0 + 255) / 256, 256, 0, stream>>>(x,    x_bf,    n0);
        cvt_bf16_kernel<<<(n1 + 255) / 256, 256, 0, stream>>>(Win,  win_bf,  n1);
        cvt_bf16_kernel<<<(n2 + 255) / 256, 256, 0, stream>>>(Wout, wout_bf, n2);
    }

    // 1) GEMM1: xp = x . Win^T ; split into u (raw) and sigmoid(g)
    gemm1_kernel<<<dim3(MROWS / 64, NP / 128), 256, 0, stream>>>(
        x_bf, win_bf, u_buf, g_buf);

    // 2) causal depthwise conv + linear-recurrence scan + gating -> bf16 y
    conv_scan_gate_kernel<<<(BB * INNERX) / 256, 256, 0, stream>>>(
        u_buf, g_buf, conv_w, state_decay, state_scale, y_bf);

    // 3) GEMM2 + fused RMSNorm
    gemm2_norm_kernel<<<MROWS / 16, 256, 0, stream>>>(
        y_bf, wout_bf, norm_w, out);
}
